// GCNNet_42777874268531
// MI455X (gfx1250) — matrix operand order repaired and checked by reading, not compiled, not run on hardware
//
#include <hip/hip_runtime.h>

typedef __attribute__((ext_vector_type(16))) _Float16 v16h;
typedef __attribute__((ext_vector_type(8)))  _Float16 v8h;
typedef __attribute__((ext_vector_type(8)))  float    v8f;

typedef __attribute__((ext_vector_type(4))) unsigned int tdm_g0_t;
typedef __attribute__((ext_vector_type(8))) int          tdm_g1_t;
typedef __attribute__((ext_vector_type(4))) int          tdm_g2_t;

// ---------------------------------------------------------------------------
// Utility: zero a float buffer (accumulators must be re-zeroed every call)
// ---------------------------------------------------------------------------
__global__ void zero_f32(float* __restrict__ p, int n) {
  int i = blockIdx.x * blockDim.x + threadIdx.x;
  if (i < n) p[i] = 0.0f;
}

// ---------------------------------------------------------------------------
// Degree via fp32 global atomics (GLOBAL_ATOMIC_ADD_F32, no-return path)
// ---------------------------------------------------------------------------
__global__ void degree_kernel(const int* __restrict__ dst, float* __restrict__ deg, int E) {
  int e = blockIdx.x * blockDim.x + threadIdx.x;
  if (e < E) atomicAdd(&deg[dst[e]], 1.0f);
}

// In-place deg -> dinv. Self loop adds +1, so deg+1 >= 1 always and the
// reference's where(deg>0, rsqrt(max(deg,1)), 0) == rsqrt(deg+1).
__global__ void dinv_kernel(float* __restrict__ deg, int N) {
  int i = blockIdx.x * blockDim.x + threadIdx.x;
  if (i < N) deg[i] = rsqrtf(deg[i] + 1.0f);
}

// ---------------------------------------------------------------------------
// WMMA GEMM: C[MxN] = act(A[MxK] @ B[KxN] + bias)
//   * block = 128 threads (4 waves); block tile = 64 rows x 64 cols
//   * A strip staged into LDS by the Tensor Data Mover (TENSORcnt), with
//     TDM row padding (+4 DWORDs/row) so LDS A rows are bank-conflict free
//   * B chunk staged in LDS as f16 in WMMA B-fragment layout (padded slots)
//   * each wave: one 16-row tile, 4 accumulators across 64 cols ->
//     each A fragment feeds 4 back-to-back v_wmma_f32_16x16x32_f16
//   * act: 0 = none, 1 = relu, 2 = leaky-relu(slope)
// Requires K in {64,128} (multiple of 32), N multiple of 64.
// ---------------------------------------------------------------------------
#define BPAD 40   // halves per (kc, n) B slot: 32 data + 8 pad -> 80B stride, 16B aligned

__global__ __launch_bounds__(128)
void gemm_wmma_f16(const float* __restrict__ A, const float* __restrict__ B,
                   const float* __restrict__ bias, float* __restrict__ C,
                   int M, int N, int K, int act, float slope) {
  __shared__ __attribute__((aligned(16))) float    lds_a[64 * 132];     // 64 rows x (K+4), K<=128
  __shared__ __attribute__((aligned(16))) _Float16 lds_b[4 * 64 * BPAD];

  const int lane = threadIdx.x & 31;
  const int wave = threadIdx.x >> 5;
  const int nl   = lane & 15;
  const int hi   = lane >> 4;          // 0: lanes 0-15, 1: lanes 16-31
  const int m0   = blockIdx.x * 64;    // block row base
  const int n1   = blockIdx.y * 64;    // block col base
  const int strideA = K + 4;           // padded floats per LDS A row (TDM pad)

  // ---- TDM async load of A[m0:m0+64, 0:K] into lds_a (wave 0 issues) ----
  if (wave == 0) {
    unsigned long long ga = (unsigned long long)(const void*)(A + (size_t)m0 * K);
    unsigned lds_base = (unsigned)(size_t)(void*)lds_a;   // LDS byte offset (addr[31:0])
    int rows = M - m0;                                    // OOB rows read as zero

    tdm_g0_t g0;
    g0[0] = 1u;                                            // count=1, load descriptor
    g0[1] = lds_base;                                      // D#.lds_addr
    g0[2] = (unsigned)(ga & 0xFFFFFFFFull);                // global_addr[31:0]
    g0[3] = (unsigned)((ga >> 32) & 0x01FFFFFFull)         // global_addr[56:32]
          | (2u << 30);                                    // type = 2 ("image")

    // group1: data_size=4B (code 2), pad_enable, pad every K DWORDs (code 6/5),
    // pad_amount = 4 DWORDs (code 3); tensor_dim0=K, tensor_dim1=rows,
    // tile_dim0=K, tile_dim1=64, tensor_dim0_stride=K.
    unsigned padi = (K == 128) ? 6u : 5u;                  // 2^(code+1) DWORDs
    unsigned long long q0 = (unsigned long long)((2u << 16) | (1u << 20) |
                                                 (padi << 22) | (3u << 25))
                          | ((unsigned long long)(unsigned)K << 48);          // dim0[15:0]
    unsigned long long q1 = (unsigned long long)(((unsigned)K >> 16) & 0xFFFF)      // dim0[31:16]
                          | (((unsigned long long)(unsigned)rows & 0xFFFFFFFFull) << 16) // dim1
                          | ((unsigned long long)((unsigned)K & 0xFFFF) << 48); // tile_dim0
    unsigned long long q2 = 64ull                                               // tile_dim1
                          | ((unsigned long long)(unsigned)K << 32);            // dim0_stride[31:0]
    tdm_g1_t g1;
    g1[0] = (int)(q0 & 0xFFFFFFFFull); g1[1] = (int)(q0 >> 32);
    g1[2] = (int)(q1 & 0xFFFFFFFFull); g1[3] = (int)(q1 >> 32);
    g1[4] = (int)(q2 & 0xFFFFFFFFull); g1[5] = (int)(q2 >> 32);
    g1[6] = 0;                         g1[7] = 0;          // dim0_stride[47:32], dim1_stride
    tdm_g2_t g2 = {};                                      // 2D tile: groups 2/3 zero
    tdm_g2_t g3 = {};
    tdm_g1_t g4 = {};                                      // extra group (clang-23 6-arg form)

    __builtin_amdgcn_tensor_load_to_lds(g0, g1, g2, g3, g4, 0);
    __builtin_amdgcn_s_wait_tensorcnt(0);
  }

  // ---- stage B chunk (K x 64) in LDS as f16 WMMA B-fragments ----
  // slot (kc, n) holds K sub-range kc*32 + [h*16 + j]; coalesced global reads.
  for (int e = threadIdx.x; e < K * 64; e += 128) {
    int n = e & 63;
    int k = e >> 6;
    float v = B[(size_t)k * N + (n1 + n)];
    int kc = k >> 5, kr = k & 31, h = kr >> 4, j = kr & 15;
    lds_b[(kc * 64 + n) * BPAD + h * 16 + j] = (_Float16)v;
  }
  __syncthreads();

  // ---- main loop: 1 A fragment -> 4 WMMAs per K-step ----
  v8f acc0 = {}, acc1 = {}, acc2 = {}, acc3 = {};
  const float* arow = lds_a + (wave * 16 + nl) * strideA;
  for (int kc = 0; kc < (K >> 5); ++kc) {
    const int k0 = kc << 5;
    v16h a;
    {
      const float* ap = arow + k0 + (hi << 3);
      const float4 f0 = *(const float4*)(ap + 0);
      const float4 f1 = *(const float4*)(ap + 4);
      const float4 f2 = *(const float4*)(ap + 16);
      const float4 f3 = *(const float4*)(ap + 20);
      a[0] = (_Float16)f0.x; a[1] = (_Float16)f0.y; a[2] = (_Float16)f0.z; a[3] = (_Float16)f0.w;
      a[4] = (_Float16)f1.x; a[5] = (_Float16)f1.y; a[6] = (_Float16)f1.z; a[7] = (_Float16)f1.w;
      a[8]  = (_Float16)f2.x; a[9]  = (_Float16)f2.y; a[10] = (_Float16)f2.z; a[11] = (_Float16)f2.w;
      a[12] = (_Float16)f3.x; a[13] = (_Float16)f3.y; a[14] = (_Float16)f3.z; a[15] = (_Float16)f3.w;
    }
    const _Float16* bb = lds_b + (size_t)kc * 64 * BPAD + hi * 8;
    #define LOAD_B(t, bv)                                                 \
      v16h bv;                                                            \
      {                                                                   \
        const _Float16* p = bb + ((t) * 16 + nl) * BPAD;                  \
        v8h lo = *(const v8h*)(p);                                        \
        v8h hv = *(const v8h*)(p + 16);                                   \
        _Pragma("unroll")                                                 \
        for (int j = 0; j < 8; ++j) { bv[j] = lo[j]; bv[8 + j] = hv[j]; } \
      }
    LOAD_B(0, b0) LOAD_B(1, b1) LOAD_B(2, b2) LOAD_B(3, b3)
    #undef LOAD_B
    acc0 = __builtin_amdgcn_wmma_f32_16x16x32_f16(false, a, false, b0, (short)0, acc0, false, false);
    acc1 = __builtin_amdgcn_wmma_f32_16x16x32_f16(false, a, false, b1, (short)0, acc1, false, false);
    acc2 = __builtin_amdgcn_wmma_f32_16x16x32_f16(false, a, false, b2, (short)0, acc2, false, false);
    acc3 = __builtin_amdgcn_wmma_f32_16x16x32_f16(false, a, false, b3, (short)0, acc3, false, false);
  }

  // ---- epilogue: bias + activation + store ----
  #pragma unroll
  for (int t = 0; t < 4; ++t) {
    const v8f acc = (t == 0) ? acc0 : (t == 1) ? acc1 : (t == 2) ? acc2 : acc3;
    const int col = n1 + t * 16 + nl;
    const float bv = bias ? bias[col] : 0.0f;
    #pragma unroll
    for (int r = 0; r < 8; ++r) {
      int m = m0 + wave * 16 + (hi << 3) + r;   // C: VGPR r -> row r (+8 for lanes>=16)
      if (m < M) {
        float v = acc[r] + bv;
        if (act == 1)      v = fmaxf(v, 0.0f);
        else if (act == 2) v = (v > 0.0f) ? v : v * slope;
        C[(size_t)m * N + col] = v;
      }
    }
  }
}

// ---------------------------------------------------------------------------
// Edge scatter: out[dst] += h[src] * dinv[src]*dinv[dst]
// 4 features per thread; feature chunk in low thread bits -> coalesced gathers.
// ---------------------------------------------------------------------------
__global__ void scatter_kernel(const float* __restrict__ h, const int* __restrict__ src,
                               const int* __restrict__ dst, const float* __restrict__ dinv,
                               float* __restrict__ out, int E, int F) {
  int tid = blockIdx.x * blockDim.x + threadIdx.x;
  int chunks = F >> 2;
  if (tid >= E * chunks) return;
  int e = tid / chunks;
  int c = (tid - e * chunks) << 2;
  int s = src[e], d = dst[e];
  float norm = dinv[s] * dinv[d];
  const float4 hv = *(const float4*)(h + (size_t)s * F + c);
  float* o = out + (size_t)d * F + c;
  atomicAdd(o + 0, hv.x * norm);
  atomicAdd(o + 1, hv.y * norm);
  atomicAdd(o + 2, hv.z * norm);
  atomicAdd(o + 3, hv.w * norm);
}

// Self-loop message + bias + relu, fused (non-atomic: one writer per element).
__global__ void finish_kernel(float* __restrict__ out, const float* __restrict__ h,
                              const float* __restrict__ dinv, const float* __restrict__ bias,
                              int N, int F) {
  int t = blockIdx.x * blockDim.x + threadIdx.x;
  if (t >= N * F) return;
  int i = t / F, f = t - i * F;
  float di = dinv[i];
  float v = out[t] + h[t] * di * di + bias[f];
  out[t] = fmaxf(v, 0.0f);
}

// ---------------------------------------------------------------------------
// Global mean pool
// ---------------------------------------------------------------------------
__global__ void pool_kernel(const float* __restrict__ a, const int* __restrict__ batch,
                            float* __restrict__ sums, float* __restrict__ cnt, int N, int F) {
  int t = blockIdx.x * blockDim.x + threadIdx.x;
  if (t >= N * F) return;
  int i = t / F, f = t - i * F;
  int g = batch[i];
  atomicAdd(&sums[(size_t)g * F + f], a[t]);
  if (f == 0) atomicAdd(&cnt[g], 1.0f);
}

__global__ void pooldiv_kernel(const float* __restrict__ sums, const float* __restrict__ cnt,
                               float* __restrict__ g0, int G, int F) {
  int t = blockIdx.x * blockDim.x + threadIdx.x;
  if (t >= G * F) return;
  int g = t / F;
  g0[t] = sums[t] / fmaxf(cnt[g], 1.0f);
}

// Final 64->1 layer + relu (too thin for WMMA; one thread per graph).
__global__ void final_kernel(const float* __restrict__ g3, const float* __restrict__ M4,
                             const float* __restrict__ c4, float* __restrict__ out,
                             int G, int Kc) {
  int g = blockIdx.x * blockDim.x + threadIdx.x;
  if (g >= G) return;
  float acc = c4[0];
  for (int k = 0; k < Kc; ++k) acc += g3[(size_t)g * Kc + k] * M4[k];
  out[g] = fmaxf(acc, 0.0f);
}

// ---------------------------------------------------------------------------
extern "C" void kernel_launch(void* const* d_in, const int* in_sizes, int n_in,
                              void* d_out, int out_size, void* d_ws, size_t ws_size,
                              hipStream_t stream) {
  const float* x   = (const float*)d_in[0];
  const int*  ei   = (const int*)d_in[1];
  const int* batch = (const int*)d_in[2];
  const float* W1  = (const float*)d_in[3];
  const float* b1  = (const float*)d_in[4];
  const float* W2  = (const float*)d_in[5];
  const float* b2  = (const float*)d_in[6];
  const float* M1  = (const float*)d_in[7];
  const float* c1  = (const float*)d_in[8];
  const float* M2  = (const float*)d_in[9];
  const float* c2  = (const float*)d_in[10];
  const float* M3  = (const float*)d_in[11];
  const float* c3  = (const float*)d_in[12];
  const float* M4  = (const float*)d_in[13];
  const float* c4  = (const float*)d_in[14];
  float* out = (float*)d_out;

  const int N   = in_sizes[2];          // 50000 nodes
  const int E   = in_sizes[1] / 2;      // 800000 edges
  const int Cin = in_sizes[0] / N;      // 128
  const int H1 = 64, H2 = 128, G = 64;

  const int* src = ei;
  const int* dst = ei + E;

  // Workspace carve-out (1 KB aligned slices of d_ws)
  char* w = (char*)d_ws;
  auto alloc = [&](size_t bytes) {
    float* p = (float*)w;
    w += (bytes + 1023) & ~(size_t)1023;
    return p;
  };
  float* dinv = alloc((size_t)N * 4);
  float* h1   = alloc((size_t)N * H1 * 4);
  float* a1   = alloc((size_t)N * H1 * 4);
  float* h2   = alloc((size_t)N * H2 * 4);
  float* a2   = alloc((size_t)N * H2 * 4);
  float* sums = alloc((size_t)G * H2 * 4);
  float* cnt  = alloc((size_t)G * 4);
  float* g0   = alloc((size_t)G * H2 * 4);
  float* g1   = alloc((size_t)G * 64 * 4);
  float* g2   = alloc((size_t)G * 64 * 4);
  float* g3   = alloc((size_t)G * 64 * 4);

  auto cdiv = [](int a, int b) { return (a + b - 1) / b; };

  // Zero the accumulators (deterministic across graph replays)
  zero_f32<<<cdiv(N, 256), 256, 0, stream>>>(dinv, N);
  zero_f32<<<cdiv(N * H1, 256), 256, 0, stream>>>(a1, N * H1);
  zero_f32<<<cdiv(N * H2, 256), 256, 0, stream>>>(a2, N * H2);
  zero_f32<<<cdiv(G * H2, 256), 256, 0, stream>>>(sums, G * H2);
  zero_f32<<<1, 256, 0, stream>>>(cnt, G);

  // Degree + symmetric normalization
  degree_kernel<<<cdiv(E, 256), 256, 0, stream>>>(dst, dinv, E);
  dinv_kernel<<<cdiv(N, 256), 256, 0, stream>>>(dinv, N);

  // GCN layer 1: h1 = x @ W1 ; aggregate ; +self-loop +bias, relu
  {
    dim3 grid(cdiv(N, 64), cdiv(H1, 64));
    gemm_wmma_f16<<<grid, 128, 0, stream>>>(x, W1, nullptr, h1, N, H1, Cin, 0, 0.0f);
  }
  scatter_kernel<<<cdiv(E * (H1 / 4), 256), 256, 0, stream>>>(h1, src, dst, dinv, a1, E, H1);
  finish_kernel<<<cdiv(N * H1, 256), 256, 0, stream>>>(a1, h1, dinv, b1, N, H1);

  // GCN layer 2
  {
    dim3 grid(cdiv(N, 64), cdiv(H2, 64));
    gemm_wmma_f16<<<grid, 128, 0, stream>>>(a1, W2, nullptr, h2, N, H2, H1, 0, 0.0f);
  }
  scatter_kernel<<<cdiv(E * (H2 / 4), 256), 256, 0, stream>>>(h2, src, dst, dinv, a2, E, H2);
  finish_kernel<<<cdiv(N * H2, 256), 256, 0, stream>>>(a2, h2, dinv, b2, N, H2);

  // Global mean pool
  pool_kernel<<<cdiv(N * H2, 256), 256, 0, stream>>>(a2, batch, sums, cnt, N, H2);
  pooldiv_kernel<<<cdiv(G * H2, 256), 256, 0, stream>>>(sums, cnt, g0, G, H2);

  // MLP head (leaky 0.2, leaky 0.1, leaky 0.1, then 64->1 relu)
  {
    dim3 grid(cdiv(G, 64), 1);
    gemm_wmma_f16<<<grid, 128, 0, stream>>>(g0, M1, c1, g1, G, 64, 128, 2, 0.2f);
    gemm_wmma_f16<<<grid, 128, 0, stream>>>(g1, M2, c2, g2, G, 64, 64, 2, 0.1f);
    gemm_wmma_f16<<<grid, 128, 0, stream>>>(g2, M3, c3, g3, G, 64, 64, 2, 0.1f);
  }
  final_kernel<<<1, 64, 0, stream>>>(g3, M4, c4, out, G, 64);

  (void)n_in; (void)out_size; (void)ws_size;
}
